// HadamardClassifier_49314814493010
// MI455X (gfx1250) — matrix operand order repaired and checked by reading, not compiled, not run on hardware
//
#include <hip/hip_runtime.h>
#include <hip/hip_bf16.h>
#include <math.h>

// ---------------------------------------------------------------------------
// HadamardClassifier: out = -(scale / (||x||_F + 1e-8)) * x @ W^T
//   x: 8192 x 2048 f32, W = proj[:1000,:2048] (Hadamard, entries +-1)
// Strategy: f16-split GEMM on v_wmma_f32_16x16x32_f16 (W exact in f16,
// x = hi+lo f16 pair), f32 accumulate; global norm via atomic reduction.
// ---------------------------------------------------------------------------

typedef __attribute__((ext_vector_type(16))) _Float16 v16h;
typedef __attribute__((ext_vector_type(8)))  float    v8f;

#define IN_CH   2048
#define OUT_CH  1000
#define NROWS   8192
#define BLK_M   128     // 8 waves * 16 rows
#define BLK_N   64      // 4 subtiles of 16
#define KSTEP   32
#define NKT     (IN_CH / KSTEP)   // 64

// ---------------- norm reduction ----------------

__global__ void hc_zero_ws(float* ws) {
    if (threadIdx.x == 0 && blockIdx.x == 0) ws[0] = 0.0f;
}

__global__ void hc_ssq(const float* __restrict__ x, float* __restrict__ ws, int n) {
    float s = 0.0f;
    for (int i = blockIdx.x * blockDim.x + threadIdx.x; i < n;
         i += gridDim.x * blockDim.x) {
        float v = x[i];
        s += v * v;
    }
    // wave32 reduction
    #pragma unroll
    for (int off = 16; off > 0; off >>= 1)
        s += __shfl_down(s, off, 32);
    __shared__ float red[8];
    int lane = threadIdx.x & 31;
    int wave = threadIdx.x >> 5;
    if (lane == 0) red[wave] = s;
    __syncthreads();
    if (threadIdx.x == 0) {
        float t = 0.0f;
        #pragma unroll
        for (int w = 0; w < 8; ++w) t += red[w];
        atomicAdd(ws, t);
    }
}

// ---------------- WMMA GEMM ----------------

// A fragment (16x32 f16, MxK) from row-major f32, split into hi/lo f16.
// ISA layout: lane<16 holds row M=lane, K = {0..7, 16..23};
//             lane>=16 holds row M=lane-16, K = {8..15, 24..31}.
__device__ __forceinline__ void hc_load_a_split(const float* __restrict__ base,
                                                int lane, v16h& hi, v16h& lo) {
    const int r   = lane & 15;
    const int sel = lane >> 4;
    const float* q = base + (size_t)r * IN_CH + sel * 8;
    float f[16];
    *reinterpret_cast<float4*>(&f[0])  = *reinterpret_cast<const float4*>(q);
    *reinterpret_cast<float4*>(&f[4])  = *reinterpret_cast<const float4*>(q + 4);
    *reinterpret_cast<float4*>(&f[8])  = *reinterpret_cast<const float4*>(q + 16);
    *reinterpret_cast<float4*>(&f[12]) = *reinterpret_cast<const float4*>(q + 20);
    #pragma unroll
    for (int i = 0; i < 16; ++i) {
        _Float16 h = (_Float16)f[i];
        hi[i] = h;
        lo[i] = (_Float16)(f[i] - (float)h);
    }
}

// B fragment (32x16 f16, KxN) from W rows (row-major over K).
// ISA layout: lane<16 holds col N=lane, K = {0..7, 16..23}; lane>=16: N=lane-16, K = {8..15, 24..31}.
// W entries are +-1 -> exact in f16, no lo term needed.
__device__ __forceinline__ v16h hc_load_b(const float* __restrict__ base, int lane) {
    const int n   = lane & 15;
    const int sel = lane >> 4;
    const float* q = base + (size_t)n * IN_CH + sel * 8;
    float f[16];
    *reinterpret_cast<float4*>(&f[0])  = *reinterpret_cast<const float4*>(q);
    *reinterpret_cast<float4*>(&f[4])  = *reinterpret_cast<const float4*>(q + 4);
    *reinterpret_cast<float4*>(&f[8])  = *reinterpret_cast<const float4*>(q + 16);
    *reinterpret_cast<float4*>(&f[12]) = *reinterpret_cast<const float4*>(q + 20);
    v16h b;
    #pragma unroll
    for (int i = 0; i < 16; ++i) b[i] = (_Float16)f[i];
    return b;
}

__global__ __launch_bounds__(256)
void hc_gemm(const float* __restrict__ x, const float* __restrict__ W,
             const float* __restrict__ scale, const float* __restrict__ ssq,
             float* __restrict__ out) {
    // double-buffered B tile in WMMA fragment layout: [buf][subtile][lane][8 dwords]
    __shared__ unsigned int ldsB[2][4][32][8];

    const int lane  = threadIdx.x & 31;
    const int wave  = threadIdx.x >> 5;
    const int nBase = blockIdx.x * BLK_N;   // 16 tiles  -> N up to 1024 (guarded)
    const int mBase = blockIdx.y * BLK_M;   // 64 tiles  -> M = 8192

    const float* aRow = x + (size_t)(mBase + wave * 16) * IN_CH;
    const float* bRow = W + (size_t)(nBase + wave * 16) * IN_CH;  // waves 0..3 only

    v8f acc[4] = {};

    for (int kt = 0; kt < NKT; ++kt) {
        const int k0  = kt * KSTEP;
        const int buf = kt & 1;

        // stage B subtile `wave` into LDS (waves 0..3)
        if (wave < 4) {
            v16h bfrag = hc_load_b(bRow + k0, lane);
            *reinterpret_cast<v16h*>(&ldsB[buf][wave][lane][0]) = bfrag;
            if (kt + 1 < NKT)
                __builtin_prefetch(bRow + (size_t)(lane & 15) * IN_CH + k0 + KSTEP, 0, 1);
        }

        // A fragments (all 8 waves, each its own 16 rows)
        v16h ahi, alo;
        hc_load_a_split(aRow + k0, lane, ahi, alo);
        if (kt + 1 < NKT)
            __builtin_prefetch(aRow + (size_t)(lane & 15) * IN_CH + k0 + KSTEP, 0, 1);

        __syncthreads();

        #pragma unroll
        for (int j = 0; j < 4; ++j) {
            v16h bfrag = *reinterpret_cast<const v16h*>(&ldsB[buf][j][lane][0]);
            acc[j] = __builtin_amdgcn_wmma_f32_16x16x32_f16(
                false, ahi, false, bfrag, (short)0, acc[j], false, false);
            acc[j] = __builtin_amdgcn_wmma_f32_16x16x32_f16(
                false, alo, false, bfrag, (short)0, acc[j], false, false);
        }
    }

    // final scaling: c = -scale / (||x|| + eps)
    const float cfac = -scale[0] / (sqrtf(ssq[0]) + 1e-8f);

    // C/D layout: vgpr r, lane -> M = r + 8*(lane>=16), N = lane%16
    const int sel = lane >> 4;
    #pragma unroll
    for (int j = 0; j < 4; ++j) {
        const int n = nBase + j * 16 + (lane & 15);
        if (n < OUT_CH) {
            #pragma unroll
            for (int r = 0; r < 8; ++r) {
                const int m = mBase + wave * 16 + r + 8 * sel;
                out[(size_t)m * OUT_CH + n] = cfac * acc[j][r];
            }
        }
    }
}

// ---------------- launcher ----------------

extern "C" void kernel_launch(void* const* d_in, const int* in_sizes, int n_in,
                              void* d_out, int out_size, void* d_ws, size_t ws_size,
                              hipStream_t stream) {
    (void)in_sizes; (void)n_in; (void)out_size; (void)ws_size;
    const float* x     = (const float*)d_in[0];
    const float* proj  = (const float*)d_in[1];   // 2048 x 2048, W = first 1000 rows
    const float* scale = (const float*)d_in[2];
    float* out = (float*)d_out;
    float* ssq = (float*)d_ws;

    hc_zero_ws<<<1, 1, 0, stream>>>(ssq);
    hc_ssq<<<2048, 256, 0, stream>>>(x, ssq, NROWS * IN_CH);

    dim3 grid((OUT_CH + BLK_N - 1) / BLK_N, NROWS / BLK_M);  // 16 x 64
    hc_gemm<<<grid, 256, 0, stream>>>(x, proj, scale, ssq, out);
}
